// NetVlad_20323785245076
// MI455X (gfx1250) — compile-verified, matrix operand order checked by reading, not run on hardware
//
#include <hip/hip_runtime.h>
#include <hip/hip_bf16.h>
#include <math.h>

// Problem constants (match reference setup_inputs)
#define N_IMG 32
#define DCH   256
#define KCL   64
#define SSP   4096   // 64*64 spatial

typedef __attribute__((ext_vector_type(16))) __bf16          v16bf;
typedef __attribute__((ext_vector_type(16))) unsigned short  v16us;
typedef __attribute__((ext_vector_type(8)))  float           v8f;
typedef __attribute__((ext_vector_type(4)))  unsigned int    v4u;

// ---- helpers -------------------------------------------------------------

__device__ __forceinline__ unsigned short f2bf(float f) {
  unsigned int u = __builtin_bit_cast(unsigned int, f);
  unsigned int r = u + 0x7FFFu + ((u >> 16) & 1u);   // round-to-nearest-even
  return (unsigned short)(r >> 16);
}

__device__ __forceinline__ float bf2f(unsigned short h) {
  unsigned int u = ((unsigned int)h) << 16;
  return __builtin_bit_cast(float, u);
}

__device__ __forceinline__ v8f wmma_bf16(v16us a, v16us b, v8f c) {
  return __builtin_amdgcn_wmma_f32_16x16x32_bf16(
      /*neg_a=*/false, __builtin_bit_cast(v16bf, a),
      /*neg_b=*/false, __builtin_bit_cast(v16bf, b),
      /*c_mod=*/(short)0, c, /*reuse_a=*/false, /*reuse_b=*/false);
}

// Generic 16x32-over-k fragment loader. Works for both operands:
//  A (16x32, row-major [m][kdim]):      row index = m    = lane%16
//  B (32x16, k-major   [n][kdim]):      row index = n    = lane%16
// Per ISA layout, lane holds kdim runs [base, base+8) and [16+base, 16+base+8)
// with base = 8*(lane/16) -> exactly two aligned 16-byte loads per lane.
__device__ __forceinline__ v16us load_frag(const unsigned short* mat,
                                           size_t ld, int lane) {
  const unsigned short* p =
      mat + (size_t)(lane & 15) * ld + (size_t)((lane >> 4) << 3);
  v4u lo = *(const v4u*)p;
  v4u hi = *(const v4u*)(p + 16);
  v16us r;
  ((v4u*)&r)[0] = lo;
  ((v4u*)&r)[1] = hi;
  return r;
}

// LDS variant (same math; lets the compiler emit ds_load_b128)
__device__ __forceinline__ v16us load_frag_lds(const unsigned short* mat,
                                               size_t ld, int lane) {
  const unsigned short* p =
      mat + (size_t)(lane & 15) * ld + (size_t)((lane >> 4) << 3);
  v4u lo = *(const v4u*)p;
  v4u hi = *(const v4u*)(p + 16);
  v16us r;
  ((v4u*)&r)[0] = lo;
  ((v4u*)&r)[1] = hi;
  return r;
}

// ---- kernel 0: conv_w -> bf16 copy, SAME [k][c] layout -------------------
// (GEMM1 B-operand is W^T, which in k-major column storage is conv_w itself.)

__global__ void k_wt(const float* __restrict__ conv_w,
                     unsigned short* __restrict__ w_bf) {
  int idx = blockIdx.x * blockDim.x + threadIdx.x;   // 16384 total
  w_bf[idx] = f2bf(conv_w[idx]);
}

// ---- kernel 1: channel L2 normalize; write xn[n][s][c] AND xnT[n][c][s] --

__global__ void k_norm(const float* __restrict__ x,
                       unsigned short* __restrict__ xn,
                       unsigned short* __restrict__ xnT) {
  int gid = blockIdx.x * blockDim.x + threadIdx.x;   // n*S threads
  int n = gid / SSP, s = gid % SSP;
  const float* xp = x + (size_t)n * DCH * SSP + s;   // stride S over channels
  float ss = 0.f;
#pragma unroll 4
  for (int c = 0; c < DCH; ++c) {
    float v = xp[(size_t)c * SSP];
    ss += v * v;
  }
  float scale = 1.0f / fmaxf(sqrtf(ss), 1e-12f);
  unsigned short* op  = xn  + ((size_t)n * SSP + s) * DCH;
  unsigned short* opT = xnT + (size_t)n * DCH * SSP + s;
#pragma unroll 4
  for (int c = 0; c < DCH; ++c) {
    unsigned short b = f2bf(xp[(size_t)c * SSP] * scale);
    op[c] = b;                        // [s][c] for GEMM1 A-operand
    opT[(size_t)c * SSP] = b;         // [c][s] for GEMM2 B-operand (coalesced)
  }
}

// ---- kernel 2: logits^T[n][s][k] = Xn * W^T + b (WMMA) -------------------
// block = (n, 64-s slab); 4 waves, each wave one 16-s M-tile, N = 64 (4 tiles)

__global__ void __launch_bounds__(128)
k_logits(const unsigned short* __restrict__ xn,
         const unsigned short* __restrict__ w_bf,
         const float* __restrict__ conv_b,
         float* __restrict__ logits) {
  __shared__ unsigned short lds_w[KCL * DCH];        // 32 KB: whole W (bf16)
  int tid = threadIdx.x;
  for (int i = tid; i < KCL * DCH; i += 128)
    ((unsigned short*)lds_w)[i] = w_bf[i];           // b32/b64 merged by compiler
  __syncthreads();

  int bn   = blockIdx.x >> 6;
  int s0   = (blockIdx.x & 63) * 64 + (tid >> 5) * 16;  // wave's M-tile
  int lane = tid & 31;

  v8f acc[4] = {};
  const unsigned short* abase = xn + ((size_t)bn * SSP + s0) * DCH;

  for (int c0 = 0; c0 < DCH; c0 += 32) {
    v16us afrag = load_frag(abase + c0, DCH, lane);        // global_load_b128
#pragma unroll
    for (int t = 0; t < 4; ++t) {
      v16us bfrag = load_frag_lds(lds_w + (size_t)(t * 16) * DCH + c0,
                                  DCH, lane);              // ds_load_b128
      acc[t] = wmma_bf16(afrag, bfrag, acc[t]);
    }
  }

  int mhi  = (lane >> 4) * 8;
  int ncol = lane & 15;
#pragma unroll
  for (int t = 0; t < 4; ++t) {
    int k = t * 16 + ncol;
    float b = conv_b[k];
#pragma unroll
    for (int r = 0; r < 8; ++r) {
      int s = s0 + r + mhi;
      logits[((size_t)bn * SSP + s) * KCL + k] = acc[t][r] + b;
    }
  }
}

// ---- kernel 3: softmax over k (contiguous), write bf16 aT[n][k][s] -------

__global__ void k_softmax(const float* __restrict__ logits,
                          unsigned short* __restrict__ aT) {
  int gid = blockIdx.x * blockDim.x + threadIdx.x;
  int n = gid / SSP, s = gid % SSP;
  const float* lp = logits + ((size_t)n * SSP + s) * KCL;
  float v[KCL];
  float m = -3.0e38f;
#pragma unroll 8
  for (int k = 0; k < KCL; ++k) { v[k] = lp[k]; m = fmaxf(m, v[k]); }
  float sum = 0.f;
#pragma unroll 8
  for (int k = 0; k < KCL; ++k) { v[k] = __expf(v[k] - m); sum += v[k]; }
  float inv = 1.0f / sum;
#pragma unroll 8
  for (int k = 0; k < KCL; ++k)
    aT[((size_t)n * KCL + k) * SSP + s] = f2bf(v[k] * inv);  // coalesced
}

// ---- kernel 4: asum[n][k] = sum_s a[n][k][s] -----------------------------

__global__ void k_asum(const unsigned short* __restrict__ aT,
                       float* __restrict__ asum) {
  int nk = blockIdx.x;                               // n*K + k
  const unsigned short* p = aT + (size_t)nk * SSP;
  __shared__ float red[256];
  float ss = 0.f;
  for (int s = threadIdx.x; s < SSP; s += 256) ss += bf2f(p[s]);
  red[threadIdx.x] = ss;
  __syncthreads();
  for (int off = 128; off > 0; off >>= 1) {
    if (threadIdx.x < off) red[threadIdx.x] += red[threadIdx.x + off];
    __syncthreads();
  }
  if (threadIdx.x == 0) asum[nk] = red[0];
}

// ---- kernel 5: vlad GEMM + centroid subtraction + intra-norm (WMMA) ------
// block = (n, 16-k tile) x all 256 c; 4 waves x 4 c-subtiles; reduce S=4096.
// No operand staging: both fragments stream from global as b128 pairs.

__global__ void __launch_bounds__(128)
k_vlad(const unsigned short* __restrict__ aT,
       const unsigned short* __restrict__ xnT,
       const float* __restrict__ asum,
       const float* __restrict__ cent,
       float* __restrict__ out) {
  __shared__ float lds_red[16 * DCH];                // vlad tile (16 KB)
  __shared__ float lds_part[128];
  __shared__ float lds_scale[16];

  int tid  = threadIdx.x;
  int bn   = blockIdx.x >> 2;
  int k0   = (blockIdx.x & 3) * 16;
  int wave = tid >> 5, lane = tid & 31;

  v8f acc[4] = {};
  const unsigned short* abase = aT + ((size_t)bn * KCL + k0) * SSP;
  const unsigned short* bbase =
      xnT + ((size_t)bn * DCH + (size_t)wave * 64) * SSP;

  for (int s0 = 0; s0 < SSP; s0 += 32) {
    if (s0 + 32 < SSP)   // stream hint: next B tile (global_prefetch_b8)
      __builtin_prefetch(bbase + s0 + 32 + (size_t)lane * SSP, 0, 1);
    v16us afrag = load_frag(abase + s0, SSP, lane);        // a[k][s] rows
#pragma unroll
    for (int t = 0; t < 4; ++t) {
      v16us bfrag = load_frag(bbase + (size_t)(t * 16) * SSP + s0,
                              SSP, lane);                  // xnT[c][s] cols
      acc[t] = wmma_bf16(afrag, bfrag, acc[t]);
    }
  }

  // subtract asum * centroid, stash f32 tile in LDS
  int mhi  = (lane >> 4) * 8;
  int ncol = lane & 15;
#pragma unroll
  for (int t = 0; t < 4; ++t) {
    int c = wave * 64 + t * 16 + ncol;
#pragma unroll
    for (int r = 0; r < 8; ++r) {
      int kl = r + mhi;
      float v = acc[t][r] -
                asum[bn * KCL + k0 + kl] * cent[(size_t)(k0 + kl) * DCH + c];
      lds_red[kl * DCH + c] = v;
    }
  }
  __syncthreads();

  // intra-cluster L2 norm over c (256 per k row)
  {
    int kl = tid >> 3, part = tid & 7;
    float ss = 0.f;
#pragma unroll
    for (int j = 0; j < 32; ++j) {
      float v = lds_red[kl * DCH + part * 32 + j];
      ss += v * v;
    }
    lds_part[tid] = ss;
  }
  __syncthreads();
  if (tid < 16) {
    float tot = 0.f;
#pragma unroll
    for (int j = 0; j < 8; ++j) tot += lds_part[tid * 8 + j];
    lds_scale[tid] = 1.0f / fmaxf(sqrtf(tot), 1e-12f);
  }
  __syncthreads();
  {
    int kl = tid >> 3, coff = (tid & 7) * 32;
    float sc = lds_scale[kl];
    float* op = out + (size_t)bn * KCL * DCH + (size_t)(k0 + kl) * DCH + coff;
#pragma unroll
    for (int j = 0; j < 32; ++j) op[j] = lds_red[kl * DCH + coff + j] * sc;
  }
}

// ---- kernel 6: global L2 norm per image (in place on d_out) --------------

__global__ void k_gnorm(float* __restrict__ out) {
  __shared__ float red[256];
  __shared__ float s_scale;
  float* p = out + (size_t)blockIdx.x * KCL * DCH;
  float ss = 0.f;
  for (int i = threadIdx.x; i < KCL * DCH; i += 256) {
    float v = p[i];
    ss += v * v;
  }
  red[threadIdx.x] = ss;
  __syncthreads();
  for (int off = 128; off > 0; off >>= 1) {
    if (threadIdx.x < off) red[threadIdx.x] += red[threadIdx.x + off];
    __syncthreads();
  }
  if (threadIdx.x == 0) s_scale = 1.0f / fmaxf(sqrtf(red[0]), 1e-12f);
  __syncthreads();
  float sc = s_scale;
  for (int i = threadIdx.x; i < KCL * DCH; i += 256) p[i] *= sc;
}

// ---- launcher ------------------------------------------------------------

extern "C" void kernel_launch(void* const* d_in, const int* in_sizes, int n_in,
                              void* d_out, int out_size, void* d_ws,
                              size_t ws_size, hipStream_t stream) {
  const float* x      = (const float*)d_in[0];   // [n, d, H, W]
  const float* conv_w = (const float*)d_in[1];   // [K, d]
  const float* conv_b = (const float*)d_in[2];   // [K]
  const float* cent   = (const float*)d_in[3];   // [K, d]
  float* out = (float*)d_out;                    // [n, K*d]

  // workspace carve-up (~176 MB total)
  unsigned short* xn  = (unsigned short*)d_ws;                // bf16 [n,S,d]
  unsigned short* xnT = xn  + (size_t)N_IMG * SSP * DCH;      // bf16 [n,d,S]
  unsigned short* aT  = xnT + (size_t)N_IMG * DCH * SSP;      // bf16 [n,K,S]
  float* logits = (float*)(aT + (size_t)N_IMG * KCL * SSP);   // f32 [n,S,K]
  float* asum   = logits + (size_t)N_IMG * SSP * KCL;         // f32 [n,K]
  unsigned short* w_bf = (unsigned short*)(asum + N_IMG * KCL); // bf16 [K,d]

  k_wt<<<dim3((KCL * DCH) / 256), dim3(256), 0, stream>>>(conv_w, w_bf);
  k_norm<<<dim3((N_IMG * SSP) / 256), dim3(256), 0, stream>>>(x, xn, xnT);
  k_logits<<<dim3(N_IMG * (SSP / 64)), dim3(128), 0, stream>>>(xn, w_bf,
                                                               conv_b, logits);
  k_softmax<<<dim3((N_IMG * SSP) / 256), dim3(256), 0, stream>>>(logits, aT);
  k_asum<<<dim3(N_IMG * KCL), dim3(256), 0, stream>>>(aT, asum);
  k_vlad<<<dim3(N_IMG * (KCL / 16)), dim3(128), 0, stream>>>(aT, xnT, asum,
                                                             cent, out);
  k_gnorm<<<dim3(N_IMG), dim3(256), 0, stream>>>(out);
}
